// PixelContrastLoss_10625749090671
// MI455X (gfx1250) — matrix operand order repaired and checked by reading, not compiled, not run on hardware
//
#include <hip/hip_runtime.h>
#include <hip/hip_bf16.h>

// ---------------------------------------------------------------------------
// PixelContrastLoss for MI455X (gfx1250, wave32, WMMA bf16 path)
//   N = 8192 rows, D = 128, binary labels, TEMP = 0.1, BASE_TEMP = 0.07
//
// z_ij = dot(f_i, f_j)/T - 10   (row max == diagonal == 1/T; the shift
//                                cancels in log_prob anyway)
// neg_i   = sum_{j: lab_j != lab_i} exp(z_ij)
// S_i     = sum_{j pos, j!=i} z_ij
// L_i     = sum_{j pos, j!=i} log(exp(z_ij) + neg_i)
// P_i     = count(lab == lab_i) - 1       (binary labels -> no pass needed)
// loss    = mean_i [ -(T/BT) * (S_i - L_i) / P_i ]
//
// Gram tiles via v_wmma_f32_16x16x32_bf16 with a 2-term bf16 split
// (AhiBhi + AhiBlo + AloBhi ~ fp32 dot products). Each wave keeps 32 A-rows
// (2 M-tiles, hi+lo) resident in VGPRs -> 24 WMMAs per 8KB of B loaded,
// halving L2 bytes/WMMA vs a 16-row tile. Epilogue is branchless.
// ---------------------------------------------------------------------------

#define NROWS   8192
#define DDIM    128
#define NTILES  (NROWS / 16)            // 512 column tiles
#define RG      32                      // rows per wave (2 M-tiles)
#define NRGRP   (NROWS / RG)            // 256 row groups
#define CSPLIT  16                      // column chunks per row group
#define CT_PER_CHUNK (NTILES / CSPLIT)  // 32 column tiles per chunk

typedef __attribute__((ext_vector_type(16))) __bf16 v16bf;
typedef __attribute__((ext_vector_type(8)))  float  v8f;

union FragU { v16bf v; uint4 q[2]; };

// A-matrix 16x32 bf16 fragment (ISA 7.12.2): lane (h,l) holds row M=l,
// K = h*8 + {0..7} and K = 16 + h*8 + {0..7}  -> two non-adjacent b128s.
__device__ __forceinline__ v16bf load_a_frag(const __bf16* __restrict__ row,
                                             int kc, int h) {
  FragU f;
  const __bf16* p = row + kc * 32 + h * 8;
  f.q[0] = *reinterpret_cast<const uint4*>(p);
  f.q[1] = *reinterpret_cast<const uint4*>(p + 16);
  return f.v;
}

// B-matrix 32x16 bf16 fragment: lane (h,n) holds column N=n,
// K = h*8 + {0..7} and K = 16 + h*8 + {0..7} -> two non-adjacent b128s.
__device__ __forceinline__ v16bf load_b_frag(const __bf16* __restrict__ row,
                                             int kc, int h) {
  FragU f;
  const __bf16* p = row + kc * 32 + h * 8;
  f.q[0] = *reinterpret_cast<const uint4*>(p);
  f.q[1] = *reinterpret_cast<const uint4*>(p + 16);
  return f.v;
}

// ---------------------------------------------------------------------------
// Kernel 1: split fp32 feats into bf16 hi/lo arrays; zero accumulators.
// ---------------------------------------------------------------------------
__global__ void __launch_bounds__(256)
prep_kernel(const float* __restrict__ feats,
            __bf16* __restrict__ hi, __bf16* __restrict__ lo,
            float* __restrict__ negAcc, float* __restrict__ posSum,
            float* __restrict__ logSum) {
  int idx = blockIdx.x * 256 + threadIdx.x;
  if (idx < NROWS * DDIM) {
    float f  = feats[idx];
    __bf16 h = (__bf16)f;
    hi[idx]  = h;
    lo[idx]  = (__bf16)(f - (float)h);
  }
  if (idx < NROWS) {
    negAcc[idx] = 0.0f; posSum[idx] = 0.0f; logSum[idx] = 0.0f;
  }
}

// ---------------------------------------------------------------------------
// Gram-tile pass body. PASS_B=false: accumulate neg / posSum.
// PASS_B=true : accumulate logSum using finished negAcc.
// One wave: 32 rows (2 M-tiles, A resident) x 512 columns (32 col tiles).
// ---------------------------------------------------------------------------
template <bool PASS_B>
__device__ __forceinline__ void gram_pass(
    const __bf16* __restrict__ hi, const __bf16* __restrict__ lo,
    const int* __restrict__ labels,
    float* __restrict__ negAcc, float* __restrict__ posSum,
    float* __restrict__ logSum) {
  const int lane = threadIdx.x & 31;
  const int wave = threadIdx.x >> 5;
  const int task = blockIdx.x * 8 + wave;          // 4096 wave-tasks
  const int rowGrp = task / CSPLIT;                // 0..255
  const int chunk  = task % CSPLIT;                // 0..15
  const int l = lane & 15;                         // fragment M / N index
  const int h = lane >> 4;                         // half-wave (K subset)
  const int rBase = rowGrp * RG;

  // A fragments (2 M-tiles, hi+lo): resident across the whole column loop.
  v16bf Ahi[2][4], Alo[2][4];
#pragma unroll
  for (int mt = 0; mt < 2; ++mt) {
    const __bf16* arow_hi = hi + (size_t)(rBase + mt * 16 + l) * DDIM;
    const __bf16* arow_lo = lo + (size_t)(rBase + mt * 16 + l) * DDIM;
#pragma unroll
    for (int kc = 0; kc < 4; ++kc) {
      Ahi[mt][kc] = load_a_frag(arow_hi, kc, h);
      Alo[mt][kc] = load_a_frag(arow_lo, kc, h);
    }
  }

  int labI[2][8];
  float negRow[2][8];
#pragma unroll
  for (int mt = 0; mt < 2; ++mt)
#pragma unroll
    for (int m = 0; m < 8; ++m) {
      const int i = rBase + mt * 16 + h * 8 + m;
      labI[mt][m]   = labels[i];
      negRow[mt][m] = PASS_B ? negAcc[i] : 0.0f;
    }

  float a0[2][8], a1[2][8];                        // per-row partials
#pragma unroll
  for (int mt = 0; mt < 2; ++mt)
#pragma unroll
    for (int m = 0; m < 8; ++m) { a0[mt][m] = 0.0f; a1[mt][m] = 0.0f; }

  const int ct0 = chunk * CT_PER_CHUNK;
  for (int ct = 0; ct < CT_PER_CHUNK; ++ct) {
    const int cBase = (ct0 + ct) * 16;
    const int j = cBase + l;                       // this lane's column
    const int labJ = labels[j];
    const __bf16* brow_hi = hi + (size_t)j * DDIM;
    const __bf16* brow_lo = lo + (size_t)j * DDIM;

    v8f acc[2];
    acc[0] = v8f{0.f, 0.f, 0.f, 0.f, 0.f, 0.f, 0.f, 0.f};
    acc[1] = v8f{0.f, 0.f, 0.f, 0.f, 0.f, 0.f, 0.f, 0.f};
#pragma unroll
    for (int kc = 0; kc < 4; ++kc) {
      v16bf Bh = load_b_frag(brow_hi, kc, h);
      v16bf Bl = load_b_frag(brow_lo, kc, h);
      // Two independent accumulator chains (one per M-tile) for WMMA ILP.
      acc[0] = __builtin_amdgcn_wmma_f32_16x16x32_bf16(false, Ahi[0][kc], false, Bh,
                                                       (short)0, acc[0], false, false);
      acc[1] = __builtin_amdgcn_wmma_f32_16x16x32_bf16(false, Ahi[1][kc], false, Bh,
                                                       (short)0, acc[1], false, false);
      acc[0] = __builtin_amdgcn_wmma_f32_16x16x32_bf16(false, Ahi[0][kc], false, Bl,
                                                       (short)0, acc[0], false, false);
      acc[1] = __builtin_amdgcn_wmma_f32_16x16x32_bf16(false, Ahi[1][kc], false, Bl,
                                                       (short)0, acc[1], false, false);
      acc[0] = __builtin_amdgcn_wmma_f32_16x16x32_bf16(false, Alo[0][kc], false, Bh,
                                                       (short)0, acc[0], false, false);
      acc[1] = __builtin_amdgcn_wmma_f32_16x16x32_bf16(false, Alo[1][kc], false, Bh,
                                                       (short)0, acc[1], false, false);
    }

    // Branchless masked accumulation: v_cmp/v_cndmask + fma, no EXEC churn.
#pragma unroll
    for (int mt = 0; mt < 2; ++mt) {
      const int iRow0 = rBase + mt * 16 + h * 8;
#pragma unroll
      for (int m = 0; m < 8; ++m) {
        float z    = acc[mt][m] * 10.0f - 10.0f;         // /TEMP, minus max
        float eqf  = (labI[mt][m] == labJ) ? 1.0f : 0.0f;
        float posf = ((iRow0 + m) == j) ? 0.0f : eqf;    // excl. diagonal
        if (PASS_B) {
          float t = __logf(__expf(z) + negRow[mt][m]);
          a0[mt][m] = fmaf(posf, t, a0[mt][m]);
        } else {
          float e = __expf(z);
          a0[mt][m] = fmaf(1.0f - eqf, e, a0[mt][m]);    // neg sum
          a1[mt][m] = fmaf(posf, z, a1[mt][m]);          // pos logit sum
        }
      }
    }
  }

  // Reduce across the 16 lanes of each half-wave (xor masks stay in-half).
#pragma unroll
  for (int mt = 0; mt < 2; ++mt)
#pragma unroll
    for (int m = 0; m < 8; ++m) {
#pragma unroll
      for (int mask = 1; mask < 16; mask <<= 1) {
        a0[mt][m] += __shfl_xor(a0[mt][m], mask, 32);
        if (!PASS_B) a1[mt][m] += __shfl_xor(a1[mt][m], mask, 32);
      }
    }
  if (l == 0) {
#pragma unroll
    for (int mt = 0; mt < 2; ++mt)
#pragma unroll
      for (int m = 0; m < 8; ++m) {
        const int i = rBase + mt * 16 + h * 8 + m;
        if (PASS_B) {
          atomicAdd(&logSum[i], a0[mt][m]);
        } else {
          atomicAdd(&negAcc[i], a0[mt][m]);
          atomicAdd(&posSum[i], a1[mt][m]);
        }
      }
  }
}

__global__ void __launch_bounds__(256)
passA_kernel(const __bf16* __restrict__ hi, const __bf16* __restrict__ lo,
             const int* __restrict__ labels, float* __restrict__ negAcc,
             float* __restrict__ posSum) {
  gram_pass<false>(hi, lo, labels, negAcc, posSum, nullptr);
}

__global__ void __launch_bounds__(256)
passB_kernel(const __bf16* __restrict__ hi, const __bf16* __restrict__ lo,
             const int* __restrict__ labels, float* __restrict__ negAcc,
             float* __restrict__ logSum) {
  gram_pass<true>(hi, lo, labels, negAcc, nullptr, logSum);
}

// ---------------------------------------------------------------------------
// Kernel 4: per-row loss, mean over rows -> scalar.
// P_i = count(lab == lab_i) - 1, computed here (binary labels).
// ---------------------------------------------------------------------------
__global__ void __launch_bounds__(256)
finalize_kernel(const float* __restrict__ posSum, const float* __restrict__ logSum,
                const int* __restrict__ labels, float* __restrict__ out) {
  __shared__ float red[256];

  // Count label==1.
  float c1 = 0.0f;
  for (int i = threadIdx.x; i < NROWS; i += 256) c1 += (float)labels[i];
  red[threadIdx.x] = c1;
  __syncthreads();
  for (int off = 128; off > 0; off >>= 1) {
    if (threadIdx.x < off) red[threadIdx.x] += red[threadIdx.x + off];
    __syncthreads();
  }
  const float cnt1 = red[0];
  __syncthreads();

  float s = 0.0f;
  for (int i = threadIdx.x; i < NROWS; i += 256) {
    float P   = (labels[i] ? cnt1 : (float)NROWS - cnt1) - 1.0f;
    float mlp = (posSum[i] - logSum[i]) / P;
    s += -(0.1f / 0.07f) * mlp;
  }
  red[threadIdx.x] = s;
  __syncthreads();
  for (int off = 128; off > 0; off >>= 1) {
    if (threadIdx.x < off) red[threadIdx.x] += red[threadIdx.x + off];
    __syncthreads();
  }
  if (threadIdx.x == 0) out[0] = red[0] * (1.0f / (float)NROWS);
}

// ---------------------------------------------------------------------------
extern "C" void kernel_launch(void* const* d_in, const int* in_sizes, int n_in,
                              void* d_out, int out_size, void* d_ws, size_t ws_size,
                              hipStream_t stream) {
  (void)in_sizes; (void)n_in; (void)out_size; (void)ws_size;
  const float* feats  = (const float*)d_in[0];   // [8192, 1, 128] fp32
  const int*   labels = (const int*)d_in[1];     // [8192] int32
  float* out = (float*)d_out;

  // Workspace layout (needs ~4.1 MB):
  __bf16* hi    = (__bf16*)d_ws;                 // 2 MB
  __bf16* lo    = hi + (size_t)NROWS * DDIM;     // 2 MB
  float* negAcc = (float*)(lo + (size_t)NROWS * DDIM);
  float* posSum = negAcc + NROWS;
  float* logSum = posSum + NROWS;

  prep_kernel<<<(NROWS * DDIM) / 256, 256, 0, stream>>>(
      feats, hi, lo, negAcc, posSum, logSum);

  const int blocks = (NRGRP * CSPLIT) / 8;       // 512 blocks x 8 waves
  passA_kernel<<<blocks, 256, 0, stream>>>(hi, lo, labels, negAcc, posSum);
  passB_kernel<<<blocks, 256, 0, stream>>>(hi, lo, labels, negAcc, logSum);
  finalize_kernel<<<1, 256, 0, stream>>>(posSum, logSum, labels, out);
}
